// Embedding_111669149789
// MI455X (gfx1250) — matrix operand order repaired and checked by reading, not compiled, not run on hardware
//
#include <hip/hip_runtime.h>

// ---------------------------------------------------------------------------
// Problem constants (from reference setup_inputs)
// ---------------------------------------------------------------------------
#define T_STEPS 4
#define BATCH   64
#define SEQ_N   196
#define PATCH   768
#define DMODEL  256
#define M_ROWS  (T_STEPS * BATCH * SEQ_N)   // 50176 = 392 * 128

#define TILE_M  128
#define TILE_N  64
#define TILE_K  32
#define NKT     (PATCH / TILE_K)            // 24

typedef __attribute__((ext_vector_type(16))) __bf16 v16bf;
typedef __attribute__((ext_vector_type(8)))  __bf16 v8bf;
typedef __attribute__((ext_vector_type(8)))  float  v8f;
typedef __attribute__((ext_vector_type(4)))  unsigned int u32x4;
typedef __attribute__((ext_vector_type(8)))  int  i32x8;
typedef __attribute__((ext_vector_type(4)))  int  i32x4;

// fp32 -> bf16, round-to-nearest-even
__device__ __forceinline__ __bf16 f2bf(float f) {
  unsigned u = __float_as_uint(f);
  u += 0x7FFFu + ((u >> 16) & 1u);
  unsigned short s = (unsigned short)(u >> 16);
  __bf16 b;
  __builtin_memcpy(&b, &s, sizeof(b));
  return b;
}

// LDS byte offset of a __shared__ object (strip the generic-AS aperture)
__device__ __forceinline__ unsigned lds_off(const void* p) {
  return (unsigned)(unsigned long long)(uintptr_t)
         (__attribute__((address_space(3))) const char*)p;
}

#define HAVE_TDM __has_builtin(__builtin_amdgcn_tensor_load_to_lds)

#if HAVE_TDM
// Issue one TDM 2-D tile load: global (row-major, 4B elements) -> LDS.
// Descriptor packing per CDNA5 ISA 8.3/8.4 (D# group0 + group1).
__device__ __forceinline__ void tdm_load_2d(unsigned lds_byte, const void* gptr,
                                            unsigned dim0, unsigned dim1,
                                            unsigned tile0, unsigned tile1,
                                            unsigned stride0) {
  unsigned long long ga = (unsigned long long)(uintptr_t)gptr;
  u32x4 g0;
  g0[0] = 1u;                                          // count=1 (valid user D#)
  g0[1] = lds_byte;                                    // lds_addr
  g0[2] = (unsigned)(ga & 0xFFFFFFFFu);                // global_addr[31:0]
  g0[3] = (unsigned)((ga >> 32) & 0x01FFFFFFu)         // global_addr[56:32]
        | (2u << 30);                                  // type=2 ("image")
  i32x8 g1;
  g1[0] = (int)(2u << 16);                             // data_size=4B; no mcast
  g1[1] = (int)((dim0 & 0xFFFFu) << 16);               // tensor_dim0[15:0]
  g1[2] = (int)(((dim0 >> 16) & 0xFFFFu)               // tensor_dim0[31:16]
        |        ((dim1 & 0xFFFFu) << 16));            // tensor_dim1[15:0]
  g1[3] = (int)(((dim1 >> 16) & 0xFFFFu)               // tensor_dim1[31:16]
        |        ((tile0 & 0xFFFFu) << 16));           // tile_dim0
  g1[4] = (int)(tile1 & 0xFFFFu);                      // tile_dim1 (tile_dim2=0)
  g1[5] = (int)stride0;                                // tensor_dim0_stride[31:0]
  g1[6] = 0;                                           // stride0 hi / stride1 lo
  g1[7] = 0;
  i32x4 z4 = {0, 0, 0, 0};
#if defined(__clang_major__) && (__clang_major__ >= 23)
  i32x8 z8 = {0, 0, 0, 0, 0, 0, 0, 0};
  __builtin_amdgcn_tensor_load_to_lds(g0, g1, z4, z4, z8, 0);
#else
  __builtin_amdgcn_tensor_load_to_lds(g0, g1, z4, z4, 0);
#endif
}
#endif // HAVE_TDM

// ---------------------------------------------------------------------------
// Kernel 1: patch-embedding GEMM  H[M,D] = X[M,K] * W[D,K]^T
//   TDM (tensor_load_to_lds) streams fp32 tiles into double-buffered LDS;
//   threads convert fp32->bf16 in LDS; waves run v_wmma_f32_16x16x32_bf16.
//   Block = 256 threads = 8 wave32; block tile 128x64; wave tile 16x64.
// ---------------------------------------------------------------------------
__global__ __launch_bounds__(256, 2)
void gemm_bf16_wmma(const float* __restrict__ X, const float* __restrict__ Wt,
                    float* __restrict__ H) {
  __shared__ __align__(16) float  fA[2][TILE_M][TILE_K];   // 2 x 16 KB (TDM dst)
  __shared__ __align__(16) float  fB[2][TILE_N][TILE_K];   // 2 x  8 KB (TDM dst)
  __shared__ __align__(32) __bf16 bA[2][TILE_M][TILE_K];   // 2 x  8 KB
  __shared__ __align__(32) __bf16 bB[2][TILE_N][TILE_K];   // 2 x  4 KB

  const int tid  = threadIdx.x;
  const int lane = tid & 31;
  const int wave = tid >> 5;                     // 0..7 -> M sub-tile
  const int rowBase = blockIdx.x * TILE_M;
  const int colBase = blockIdx.y * TILE_N;

  v8f acc0 = {}, acc1 = {}, acc2 = {}, acc3 = {};

  // conversion-pass work split
  const int ar = tid >> 1;              // A row 0..127
  const int as = (tid & 1) * 16;        // A k-seg {0,16}
  const int bc = tid >> 2;              // B col 0..63
  const int bs = (tid & 3) * 8;         // B k-seg {0,8,16,24}

#if HAVE_TDM
  const unsigned ldsA0 = lds_off(&fA[0][0][0]), ldsA1 = lds_off(&fA[1][0][0]);
  const unsigned ldsB0 = lds_off(&fB[0][0][0]), ldsB1 = lds_off(&fB[1][0][0]);
  if (wave == 0) {     // prologue: fetch k-tile 0 into buffer 0
    tdm_load_2d(ldsA0, X  + (size_t)rowBase * PATCH, PATCH, M_ROWS, TILE_K, TILE_M, PATCH);
    tdm_load_2d(ldsB0, Wt + (size_t)colBase * PATCH, PATCH, DMODEL, TILE_K, TILE_N, PATCH);
  }
#endif

  for (int it = 0; it < NKT; ++it) {
    const int p  = it & 1;
    const int kt = it * TILE_K;

#if HAVE_TDM
    if (wave == 0) __builtin_amdgcn_s_wait_tensorcnt(0);  // fp32 buf[p] landed
    __syncthreads();                                      // publish to all waves
    if (wave == 0 && it + 1 < NKT) {                      // async-prefetch next tile
      const unsigned la = p ? ldsA0 : ldsA1;
      const unsigned lb = p ? ldsB0 : ldsB1;
      tdm_load_2d(la, X  + (size_t)rowBase * PATCH + (kt + TILE_K), PATCH, M_ROWS,
                  TILE_K, TILE_M, PATCH);
      tdm_load_2d(lb, Wt + (size_t)colBase * PATCH + (kt + TILE_K), PATCH, DMODEL,
                  TILE_K, TILE_N, PATCH);
    }
#else
    { // fallback: VGPR staging straight into the fp32 buffers
      const float* ag = X  + (size_t)(rowBase + ar) * PATCH + kt + as;
      const float* bg = Wt + (size_t)(colBase + bc) * PATCH + kt + bs;
      #pragma unroll
      for (int i = 0; i < 16; ++i) fA[p][ar][as + i] = ag[i];
      #pragma unroll
      for (int i = 0; i < 8; ++i)  fB[p][bc][bs + i] = bg[i];
      __syncthreads();
    }
#endif

    // fp32 -> bf16 conversion pass (LDS -> LDS)
    #pragma unroll
    for (int i = 0; i < 16; ++i) bA[p][ar][as + i] = f2bf(fA[p][ar][as + i]);
    #pragma unroll
    for (int i = 0; i < 8; ++i)  bB[p][bc][bs + i] = f2bf(fB[p][bc][bs + i]);
    __syncthreads();

    // A fragment 16x32 bf16: lanes 0-15 K{0..15},
    // lanes 16-31 K{16..31}, row = lane%16.
    const int arow = wave * 16 + (lane & 15);
    const int ak0  = (lane < 16) ? 0 : 16;
    v8bf alo = *(const v8bf*)&bA[p][arow][ak0];
    v8bf ahi = *(const v8bf*)&bA[p][arow][ak0 + 8];
    v16bf afrag;
    #pragma unroll
    for (int i = 0; i < 8; ++i) { afrag[i] = alo[i]; afrag[i + 8] = ahi[i]; }

    // B fragments 32x16 bf16: lane = column, contiguous K half-range.
    const int bk = (lane < 16) ? 0 : 16;
    const int bl = lane & 15;
    v16bf b0 = *(const v16bf*)&bB[p][ 0 + bl][bk];
    v16bf b1 = *(const v16bf*)&bB[p][16 + bl][bk];
    v16bf b2 = *(const v16bf*)&bB[p][32 + bl][bk];
    v16bf b3 = *(const v16bf*)&bB[p][48 + bl][bk];

    acc0 = __builtin_amdgcn_wmma_f32_16x16x32_bf16(false, afrag, false, b0,
                                                   (short)0, acc0, false, false);
    acc1 = __builtin_amdgcn_wmma_f32_16x16x32_bf16(false, afrag, false, b1,
                                                   (short)0, acc1, false, false);
    acc2 = __builtin_amdgcn_wmma_f32_16x16x32_bf16(false, afrag, false, b2,
                                                   (short)0, acc2, false, false);
    acc3 = __builtin_amdgcn_wmma_f32_16x16x32_bf16(false, afrag, false, b3,
                                                   (short)0, acc3, false, false);
    __syncthreads();
  }

  // Epilogue per ISA C/D layout: VGPR i -> M=i (lanes 0-15) or i+8 (16-31),
  // N = lane%16. H laid out [T,B,N,D] == [M_ROWS, DMODEL].
  const int orow0 = rowBase + wave * 16 + ((lane < 16) ? 0 : 8);
  const int ocol  = colBase + (lane & 15);
  #pragma unroll
  for (int i = 0; i < 8; ++i) {
    size_t r = (size_t)(orow0 + i) * DMODEL + ocol;
    H[r +  0] = acc0[i];
    H[r + 16] = acc1[i];
    H[r + 32] = acc2[i];
    H[r + 48] = acc3[i];
  }
}

// ---------------------------------------------------------------------------
// Kernel 2a: zero BN-stat accumulators (every launch; ws not re-poisoned).
// ---------------------------------------------------------------------------
__global__ void bn_stats_init(float* __restrict__ stats) {
  int i = blockIdx.x * blockDim.x + threadIdx.x;
  if (i < 2 * DMODEL) stats[i] = 0.0f;
}

// Kernel 2b: per-channel sum / sum-of-squares over all M_ROWS rows.
__global__ __launch_bounds__(256)
void bn_stats(const float* __restrict__ H, float* __restrict__ stats) {
  const int d    = threadIdx.x;            // channel 0..255 (coalesced per row)
  const int row0 = blockIdx.x * 128;
  float s = 0.f, ss = 0.f;
  for (int r = 0; r < 128; ++r) {
    float v = H[(size_t)(row0 + r) * DMODEL + d];
    s  += v;
    ss += v * v;
  }
  atomicAdd(&stats[d], s);
  atomicAdd(&stats[DMODEL + d], ss);
}

// ---------------------------------------------------------------------------
// Kernel 3: fused BN-normalize + LIF1 + tAPE + LIF2, in place on H.
// One thread per 4 consecutive channels (float4 b128 traffic); T=4 in regs.
// ---------------------------------------------------------------------------
__global__ __launch_bounds__(256)
void bn_lif_pe_lif(float* __restrict__ H, const float* __restrict__ stats,
                   const float* __restrict__ gamma, const float* __restrict__ beta) {
  const int idx  = blockIdx.x * blockDim.x + threadIdx.x;   // over B*N*D/4
  const int e0   = idx * 4;
  const int d0   = e0 & (DMODEL - 1);
  const int n    = (e0 / DMODEL) % SEQ_N;

  const float invM = 1.0f / (float)M_ROWS;
  float scl[4], sft[4], pe[4];
  #pragma unroll
  for (int c = 0; c < 4; ++c) {
    const int d    = d0 + c;
    const float mean = stats[d] * invM;
    const float var  = stats[DMODEL + d] * invM - mean * mean;
    scl[c] = gamma[d] * rsqrtf(var + 1e-5f);
    sft[c] = beta[d] - mean * scl[c];
    const int   j2  = d & ~1;
    const float dv  = __expf((float)j2 * (-9.21034037e+0f / (float)DMODEL));
    const float arg = (float)n * dv * ((float)DMODEL / (float)SEQ_N);
    pe[c] = (d & 1) ? __cosf(arg) : __sinf(arg);
  }

  const size_t stride = (size_t)BATCH * SEQ_N * DMODEL;
  float v1[4] = {0.f, 0.f, 0.f, 0.f}, v2[4] = {0.f, 0.f, 0.f, 0.f};
  #pragma unroll
  for (int t = 0; t < T_STEPS; ++t) {
    float4* ptr = (float4*)(H + (size_t)t * stride + e0);
    float4 x4 = *ptr;
    float xv[4] = {x4.x, x4.y, x4.z, x4.w};
    #pragma unroll
    for (int c = 0; c < 4; ++c) {
      const float xt = xv[c] * scl[c] + sft[c];
      v1[c] = 0.5f * (v1[c] + xt);                       // LIF1, tau=2
      const float s1 = (v1[c] >= 0.5f) ? 1.f : 0.f;
      v1[c] *= (1.f - s1);                               // hard reset
      v2[c] = 0.5f * (v2[c] + (s1 + pe[c]));             // +PE, LIF2
      const float s2 = (v2[c] >= 0.5f) ? 1.f : 0.f;
      v2[c] *= (1.f - s2);
      xv[c] = s2;
    }
    *ptr = make_float4(xv[0], xv[1], xv[2], xv[3]);      // [T,B,N,D] output
  }
}

// ---------------------------------------------------------------------------
extern "C" void kernel_launch(void* const* d_in, const int* in_sizes, int n_in,
                              void* d_out, int out_size, void* d_ws, size_t ws_size,
                              hipStream_t stream) {
  const float* x     = (const float*)d_in[0];  // [T,B,N,P]
  const float* W     = (const float*)d_in[1];  // [D,P]
  const float* gamma = (const float*)d_in[2];  // [D]
  const float* beta  = (const float*)d_in[3];  // [D]
  float* H     = (float*)d_out;                // [T,B,N,D] (also GEMM output)
  float* stats = (float*)d_ws;                 // 2*D floats

  dim3 ggrid(M_ROWS / TILE_M, DMODEL / TILE_N);      // 392 x 4
  gemm_bf16_wmma<<<ggrid, 256, 0, stream>>>(x, W, H);

  bn_stats_init<<<1, 512, 0, stream>>>(stats);
  bn_stats<<<M_ROWS / 128, 256, 0, stream>>>(H, stats);

  const int total4 = BATCH * SEQ_N * DMODEL / 4;     // 802816 = 3136 * 256
  bn_lif_pe_lif<<<total4 / 256, 256, 0, stream>>>(H, stats, gamma, beta);
}
